// CrossModalAttentionLayer_47682726920245
// MI455X (gfx1250) — matrix operand order repaired and checked by reading, not compiled, not run on hardware
//
#include <hip/hip_runtime.h>

// Problem sizes (fixed by the reference)
#define BSZ   64
#define NSP   1024
#define XCH   1024
#define KOUT  256
#define YSZ   768

#define TILE_M 128          // rows per block (8 waves x 16 rows)
#define WROW   40           // LDS row stride in bf16 (80B: conflict-free for 16-lane column reads)

typedef __attribute__((ext_vector_type(16))) __bf16 v16bf;
typedef __attribute__((ext_vector_type(8)))  __bf16 v8bf;
typedef __attribute__((ext_vector_type(8)))  float  v8f;

union V16U { v16bf v; v8bf h[2]; };

// Hardware transcendental tanh (gfx1250 V_TANH_F32). The trailing v_nop covers
// the documented TRANS-op hazard (1 independent op before the result is used).
__device__ __forceinline__ float tanh_hw(float x) {
  float r;
  asm volatile("v_tanh_f32 %0, %1\n\tv_nop" : "=v"(r) : "v"(x));
  return r;
}

// ---------------------------------------------------------------------------
// Kernel 1: c[b,k] = b_ch[k] + sum_s y[b,s] * W_y[k,s]   (64x256, trivial work)
// ---------------------------------------------------------------------------
__global__ __launch_bounds__(256) void yk_bias_kernel(
    const float* __restrict__ y, const float* __restrict__ W_y,
    const float* __restrict__ b_ch, float* __restrict__ cbuf) {
  const int b = blockIdx.x;
  const int k = threadIdx.x;
  const float* yr = y + (size_t)b * YSZ;
  const float* wr = W_y + (size_t)k * YSZ;
  float acc = 0.f;
#pragma unroll 4
  for (int s = 0; s < YSZ; s += 4) {
    float4 a = *(const float4*)(yr + s);
    float4 w = *(const float4*)(wr + s);
    acc += a.x * w.x + a.y * w.y + a.z * w.z + a.w * w.w;
  }
  cbuf[b * KOUT + k] = acc + b_ch[k];
}

// ---------------------------------------------------------------------------
// Kernel 2: fused  out = softmax_k( tanh( x @ W_ch^T + c ) )
// One wave: 16 rows x 256 channels, accumulated in 16 bf16-WMMA tiles.
// Double-buffered LDS for the W_ch k-chunk: one barrier per chunk, next
// chunk's global loads overlap the 16 WMMAs.
// ---------------------------------------------------------------------------
__global__ __launch_bounds__(256) void fused_attn_kernel(
    const float* __restrict__ x, const float* __restrict__ W_ch,
    const float* __restrict__ cbuf, float* __restrict__ out) {
  __shared__ __bf16 ldsW[2][KOUT * WROW];   // 2 x 20 KB: W_ch[:, kc:kc+32] as bf16

  const int b     = blockIdx.y;
  const int mtile = blockIdx.x;
  const int tid   = threadIdx.x;
  const int lane  = tid & 31;
  const int wave  = tid >> 5;
  const int half  = lane >> 4;   // lane group (0: lanes 0-15, 1: lanes 16-31)
  const int n     = lane & 15;   // column within 16-wide tile / A-matrix row

  // A-matrix row for this lane (A layout: row = lane%16 for both halves)
  const int rowA = mtile * TILE_M + wave * 16 + n;
  const float* __restrict__ xrow = x + ((size_t)b * NSP + rowA) * XCH;
  const float* __restrict__ wrow = W_ch + (size_t)tid * XCH;   // tid = out-channel row it stages

  v8f acc[16];
#pragma unroll
  for (int t = 0; t < 16; ++t)
#pragma unroll
    for (int r = 0; r < 8; ++r) acc[t][r] = 0.f;

  // ---- prologue: stage chunk 0 into buffer 0
  {
    float wv[32];
#pragma unroll
    for (int j = 0; j < 32; j += 4) {
      float4 w = *(const float4*)(wrow + j);
      wv[j] = w.x; wv[j + 1] = w.y; wv[j + 2] = w.z; wv[j + 3] = w.w;
    }
    __bf16* dst = &ldsW[0][tid * WROW];   // byte offset 80*tid -> 16B aligned
#pragma unroll
    for (int q = 0; q < 4; ++q) {
      v8bf p;
#pragma unroll
      for (int j = 0; j < 8; ++j) p[j] = (__bf16)wv[q * 8 + j];
      *(v8bf*)(dst + q * 8) = p;
    }
  }
  __syncthreads();

  for (int kc = 0; kc < XCH; kc += 32) {
    const int  cur  = (kc >> 5) & 1;
    const bool more = (kc + 32) < XCH;

    // ---- issue next chunk's W_ch global loads early (hidden under the WMMAs)
    float wv[32];
    if (more) {
#pragma unroll
      for (int j = 0; j < 32; j += 4) {
        float4 w = *(const float4*)(wrow + kc + 32 + j);
        wv[j] = w.x; wv[j + 1] = w.y; wv[j + 2] = w.z; wv[j + 3] = w.w;
      }
    }

    // ---- A fragment (16x32 bf16): lane holds K = kc + 8*half + [0..8) and kc+16+8*half+[0..8)
    const float* ap = xrow + kc + half * 8;
    float4 f0 = *(const float4*)(ap);
    float4 f1 = *(const float4*)(ap + 4);
    float4 f2 = *(const float4*)(ap + 16);
    float4 f3 = *(const float4*)(ap + 20);
    __builtin_prefetch(ap + 32, 0, 1);   // global_prefetch_b8: next k-chunk of x

    v16bf a;
    a[0]  = (__bf16)f0.x; a[1]  = (__bf16)f0.y; a[2]  = (__bf16)f0.z; a[3]  = (__bf16)f0.w;
    a[4]  = (__bf16)f1.x; a[5]  = (__bf16)f1.y; a[6]  = (__bf16)f1.z; a[7]  = (__bf16)f1.w;
    a[8]  = (__bf16)f2.x; a[9]  = (__bf16)f2.y; a[10] = (__bf16)f2.z; a[11] = (__bf16)f2.w;
    a[12] = (__bf16)f3.x; a[13] = (__bf16)f3.y; a[14] = (__bf16)f3.z; a[15] = (__bf16)f3.w;

    // ---- 16 N-tiles: B[cr, col] = W_ch[t*16+n, kc+cr]; lane holds cr = 8*half + [0..8) and 16+8*half+[0..8)
#pragma unroll
    for (int t = 0; t < 16; ++t) {
      const __bf16* bp = &ldsW[cur][(t * 16 + n) * WROW + half * 8];  // 16B-aligned
      V16U u;
      u.h[0] = *(const v8bf*)(bp);
      u.h[1] = *(const v8bf*)(bp + 16);
      acc[t] = __builtin_amdgcn_wmma_f32_16x16x32_bf16(
          false, a, false, u.v, (short)0, acc[t], false, false);
    }

    // ---- convert + store next chunk into the other buffer
    if (more) {
      __bf16* dst = &ldsW[cur ^ 1][tid * WROW];
#pragma unroll
      for (int q = 0; q < 4; ++q) {
        v8bf p;
#pragma unroll
        for (int j = 0; j < 8; ++j) p[j] = (__bf16)wv[q * 8 + j];
        *(v8bf*)(dst + q * 8) = p;
      }
    }
    __syncthreads();
  }

  // ---- epilogue: +c, tanh, softmax over 256 channels, store
  // D layout: lane(half,n), VGPR r -> D[row = 8*half + r, col = 16*t + n]
  const float* crow = cbuf + b * KOUT;
  float cv[16];
#pragma unroll
  for (int t = 0; t < 16; ++t) cv[t] = crow[t * 16 + n];

  float* obase = out +
      ((size_t)b * NSP + mtile * TILE_M + wave * 16 + half * 8) * KOUT;

#pragma unroll
  for (int r = 0; r < 8; ++r) {
    float v[16];
    float mx = -3.402823466e38f;
#pragma unroll
    for (int t = 0; t < 16; ++t) {
      v[t] = tanh_hw(acc[t][r] + cv[t]);   // hardware v_tanh_f32
      mx = fmaxf(mx, v[t]);
    }
    // reduce over the 16 lanes holding this row (width-16 groups)
#pragma unroll
    for (int off = 1; off < 16; off <<= 1)
      mx = fmaxf(mx, __shfl_xor(mx, off, 16));
    float s = 0.f;
#pragma unroll
    for (int t = 0; t < 16; ++t) {
      v[t] = __expf(v[t] - mx);            // v_exp_f32
      s += v[t];
    }
#pragma unroll
    for (int off = 1; off < 16; off <<= 1)
      s += __shfl_xor(s, off, 16);
    const float inv = __builtin_amdgcn_rcpf(s);   // v_rcp_f32 (softmax norm)
    float* orow = obase + (size_t)r * KOUT;
#pragma unroll
    for (int t = 0; t < 16; ++t) orow[t * 16 + n] = v[t] * inv;
  }
}

// ---------------------------------------------------------------------------
extern "C" void kernel_launch(void* const* d_in, const int* in_sizes, int n_in,
                              void* d_out, int out_size, void* d_ws, size_t ws_size,
                              hipStream_t stream) {
  const float* x    = (const float*)d_in[0];
  const float* y    = (const float*)d_in[1];
  const float* W_ch = (const float*)d_in[2];
  const float* b_ch = (const float*)d_in[3];
  const float* W_y  = (const float*)d_in[4];
  float* out  = (float*)d_out;
  float* cbuf = (float*)d_ws;   // 64*256 floats = 64 KB scratch

  yk_bias_kernel<<<BSZ, KOUT, 0, stream>>>(y, W_y, b_ch, cbuf);
  fused_attn_kernel<<<dim3(NSP / TILE_M, BSZ), 256, 0, stream>>>(x, W_ch, cbuf, out);
}
